// TGAT_16131897164049
// MI455X (gfx1250) — matrix operand order repaired and checked by reading, not compiled, not run on hardware
//
#include <hip/hip_runtime.h>
#include <hip/hip_bf16.h>

typedef __attribute__((ext_vector_type(16))) __bf16 bf16x16;
typedef __attribute__((ext_vector_type(8)))  float  f32x8;

#define NNODE 1000
#define NBATCH 32
#define NTOT  (NNODE*NBATCH)     // 32000
#define CC 32
#define NH 8
#define NE 384000
#define H1D 32
#define H2D 128
#define NPRED 2
#define NEG_SLOPE 0.2f

// ---- WMMA fragment index helpers (contiguous k = 16*hi + p, same for A and B; wave32, 16-bit operands) ----
__device__ __forceinline__ int akidx(int hi, int p) { return 16*hi + p; }
__device__ __forceinline__ int bkidx(int hi, int p) { return 16*hi + p; }

// monotone float<->uint for atomicMax-based segment max
__device__ __forceinline__ unsigned fenc(float f) {
  unsigned u = __float_as_uint(f);
  return (u & 0x80000000u) ? ~u : (u | 0x80000000u);
}
__device__ __forceinline__ float fdec(unsigned u) {
  unsigned b = (u & 0x80000000u) ? (u & 0x7fffffffu) : ~u;
  return __uint_as_float(b);
}
__device__ __forceinline__ float sigmoidf(float x) { return 1.f / (1.f + __expf(-x)); }

// ---------------- init: gat accumulator = bias, segment max = -inf, denom = 0
__global__ void k_init(float* __restrict__ gat_out, const float* __restrict__ b_gat,
                       unsigned* __restrict__ menc, float* __restrict__ denom) {
  int i = blockIdx.x * blockDim.x + threadIdx.x;
  if (i < NTOT * CC) gat_out[i] = b_gat[i & (CC - 1)];
  if (i < NTOT * NH) { menc[i] = 0x007FFFFFu; denom[i] = 0.f; }   // enc(-inf)
}

// ---------------- x_lin = x @ W_gat   (M=32000, N=256, K=32, one WMMA per tile)
__global__ void k_gat_gemm(const float* __restrict__ x, const float* __restrict__ Wg,
                           float* __restrict__ xlin) {
  int lane = threadIdx.x & 31, warp = threadIdx.x >> 5;
  int gw = blockIdx.x * 8 + warp;            // 32000 waves
  int mt = gw >> 4, nt = gw & 15;
  int hi = lane >> 4, ln = lane & 15;
  int row = mt * 16 + ln;
  int col = nt * 16 + ln;
  bf16x16 a, b;
#pragma unroll
  for (int p = 0; p < 16; ++p) {
    a[p] = (__bf16)x[row * CC + akidx(hi, p)];
    b[p] = (__bf16)Wg[bkidx(hi, p) * (NH * CC) + col];
  }
  f32x8 c = {0.f, 0.f, 0.f, 0.f, 0.f, 0.f, 0.f, 0.f};
  c = __builtin_amdgcn_wmma_f32_16x16x32_bf16(false, a, false, b, (short)0, c, false, false);
#pragma unroll
  for (int r = 0; r < 8; ++r) {
    int m = mt * 16 + r + 8 * hi;
    xlin[(long)m * (NH * CC) + col] = c[r];
  }
}

// ---------------- per-node attention coefficients asrc/adst [N, H]
__global__ void k_attn(const float* __restrict__ xlin, const float* __restrict__ a_src,
                       const float* __restrict__ a_dst, float* __restrict__ asrc,
                       float* __restrict__ adst) {
  int i = blockIdx.x * blockDim.x + threadIdx.x;
  if (i >= NTOT * NH) return;
  int n = i >> 3, h = i & 7;
  const float* xr = xlin + (long)n * (NH * CC) + h * CC;
  const float* as = a_src + h * CC;
  const float* ad = a_dst + h * CC;
  float s = 0.f, d = 0.f;
#pragma unroll
  for (int cdx = 0; cdx < CC; ++cdx) { float v = xr[cdx]; s += v * as[cdx]; d += v * ad[cdx]; }
  asrc[i] = s; adst[i] = d;
}

// ---------------- per-edge (all heads): leaky_relu score + segment max
__global__ void k_edge_score(const int* __restrict__ src, const int* __restrict__ dst,
                             const float* __restrict__ asrc, const float* __restrict__ adst,
                             float* __restrict__ esc, unsigned* __restrict__ menc) {
  int e = blockIdx.x * blockDim.x + threadIdx.x;
  if (e >= NE) return;
  int s = src[e], d = dst[e];
#pragma unroll
  for (int h = 0; h < NH; ++h) {
    float v = asrc[s * NH + h] + adst[d * NH + h];
    v = (v > 0.f) ? v : NEG_SLOPE * v;               // leaky_relu
    esc[e * NH + h] = v;
    atomicMax(&menc[d * NH + h], fenc(v));
  }
}

// ---------------- per-edge: ew = exp(e - m[dst]);  denom = segment_sum(ew)
__global__ void k_edge_denom(const int* __restrict__ dst, const unsigned* __restrict__ menc,
                             float* __restrict__ esc, float* __restrict__ denom) {
  int e = blockIdx.x * blockDim.x + threadIdx.x;
  if (e >= NE) return;
  int d = dst[e];
#pragma unroll
  for (int h = 0; h < NH; ++h) {
    float m = fdec(menc[d * NH + h]);
    float ew = __expf(esc[e * NH + h] - m);
    esc[e * NH + h] = ew;
    atomicAdd(&denom[d * NH + h], ew);
  }
}

// ---------------- per-edge scatter with fused head-mean:
// gat_out[dst,c] += sum_h (alpha_h/H) * x_lin[src,h,c]  -> 32 atomics/edge (8x fewer)
__global__ void k_edge_agg(const int* __restrict__ src, const int* __restrict__ dst,
                           const float* __restrict__ esc, const float* __restrict__ denom,
                           const float* __restrict__ xlin, float* __restrict__ gat_out) {
  int e = blockIdx.x * blockDim.x + threadIdx.x;
  if (e >= NE) return;
  int s = src[e], d = dst[e];
  float w[NH];
#pragma unroll
  for (int h = 0; h < NH; ++h)
    w[h] = esc[e * NH + h] / fmaxf(denom[d * NH + h], 1e-16f) * (1.f / NH);
  float acc[CC];
#pragma unroll
  for (int cdx = 0; cdx < CC; ++cdx) acc[cdx] = 0.f;
  const float* xr = xlin + (long)s * (NH * CC);      // 1KB contiguous -> b128 loads
#pragma unroll
  for (int h = 0; h < NH; ++h)
#pragma unroll
    for (int cdx = 0; cdx < CC; ++cdx) acc[cdx] += w[h] * xr[h * CC + cdx];
  float* outr = gat_out + d * CC;
#pragma unroll
  for (int cdx = 0; cdx < CC; ++cdx) atomicAdd(&outr[cdx], acc[cdx]);
}

// ---------------- hoisted LSTM1 input projection: G1[(t,b), 0..127] = seq @ W_ih1^T + biases
// seq[t,b,j] = gat_out[(b*1000+j)*32 + t];  M=1024, N=128, K=1000 (zero-padded to 1024)
// Each block = one M-tile, 8 waves = 8 N-tiles. Each wave double-buffers its own
// 16-col x 32-K B stripe in LDS via global_load_async_to_lds_b128 (coalesced along K),
// synchronized per-wave with s_wait_asynccnt (no cross-wave barriers needed).
__global__ void k_lstm_pre(const float* __restrict__ gat_out, const float* __restrict__ Wih1,
                           const float* __restrict__ bih1, const float* __restrict__ bhh1,
                           float* __restrict__ G1) {
  __shared__ float Bw[8][2][16 * 32];                // 32 KB: [wave][buf][col][k]
  int lane = threadIdx.x & 31, warp = threadIdx.x >> 5;
  int mt = blockIdx.x;                               // 64 blocks
  int nt = warp;                                     // 0..7
  int hi = lane >> 4, ln = lane & 15;
  int m = mt * 16 + ln;                              // row = t*32 + b
  int t = m >> 5, bb = m & 31;
  int col = nt * 16 + ln;                            // 0..127
  unsigned lds_base = (unsigned)(unsigned long long)(&Bw[warp][0][0]);

  auto stage = [&](int kt, int bf) {
    if (kt < 31) {                                   // fully in-bounds: async DMA to LDS
#pragma unroll
      for (int q = 0; q < 4; ++q) {
        int cnk = lane + 32 * q;                     // 128 chunks of 16B
        int cl = cnk >> 3, kg = cnk & 7;
        int gcol = nt * 16 + cl;
        unsigned long long gaddr =
            (unsigned long long)(Wih1 + (size_t)gcol * NNODE + kt * 32 + kg * 4);
        unsigned laddr = lds_base + (unsigned)(bf * (16 * 32) + cl * 32 + kg * 4) * 4u;
        asm volatile("global_load_async_to_lds_b128 %0, %1, off"
                     :: "v"(laddr), "v"(gaddr) : "memory");
      }
    } else {                                         // ragged tail (j >= 1000 -> 0)
#pragma unroll
      for (int q = 0; q < 4; ++q) {
        int cnk = lane + 32 * q;
        int cl = cnk >> 3, kg = cnk & 7;
        int gcol = nt * 16 + cl;
#pragma unroll
        for (int u = 0; u < 4; ++u) {
          int j = kt * 32 + kg * 4 + u;
          Bw[warp][bf][cl * 32 + kg * 4 + u] =
              (j < NNODE) ? Wih1[(size_t)gcol * NNODE + j] : 0.f;
        }
      }
    }
  };

  stage(0, 0);
  f32x8 c = {0.f, 0.f, 0.f, 0.f, 0.f, 0.f, 0.f, 0.f};
  for (int kt = 0; kt < 32; ++kt) {
    int bf = kt & 1;
    if (kt < 31) stage(kt + 1, bf ^ 1);
    if (kt < 30) asm volatile("s_wait_asynccnt 0x4" ::: "memory");  // tile kt done, kt+1 in flight
    else         asm volatile("s_wait_asynccnt 0x0" ::: "memory");  // drain (tail is DS, in-order)
    bf16x16 a, b;
#pragma unroll
    for (int p = 0; p < 16; ++p) {
      int ja = kt * 32 + akidx(hi, p);
      a[p] = (__bf16)((ja < NNODE) ? gat_out[(long)(bb * NNODE + ja) * CC + t] : 0.f);
      b[p] = (__bf16)Bw[warp][bf][ln * 32 + bkidx(hi, p)];
    }
    c = __builtin_amdgcn_wmma_f32_16x16x32_bf16(false, a, false, b, (short)0, c, false, false);
  }
  float bias = bih1[col] + bhh1[col];
#pragma unroll
  for (int r = 0; r < 8; ++r) {
    int mo = mt * 16 + r + 8 * hi;
    G1[mo * (4 * H1D) + col] = c[r] + bias;
  }
}

// ---------------- fused stacked LSTM recurrence, single persistent workgroup (32 waves)
__global__ void __launch_bounds__(1024)
k_lstm_rec(const float* __restrict__ G1, const float* __restrict__ Whh1,
           const float* __restrict__ Wih2, const float* __restrict__ Whh2,
           const float* __restrict__ bih2, const float* __restrict__ bhh2,
           float* __restrict__ h2out) {
  __shared__ __bf16 h1s[NBATCH * H1D];       // 2 KB
  __shared__ float  c1s[NBATCH * H1D];       // 4 KB
  __shared__ __bf16 h2s[NBATCH * H2D];       // 8 KB
  __shared__ float  c2s[NBATCH * H2D];       // 16 KB
  __shared__ __bf16 g2s[NBATCH * 4 * H2D];   // 32 KB (gates2; gates1 reuses first 8 KB)
  int tid = threadIdx.x;
  int lane = tid & 31, warp = tid >> 5;
  int hi = lane >> 4, ln = lane & 15;
  for (int s = tid; s < NBATCH * H1D; s += 1024) { h1s[s] = (__bf16)0.f; c1s[s] = 0.f; }
  for (int s = tid; s < NBATCH * H2D; s += 1024) { h2s[s] = (__bf16)0.f; c2s[s] = 0.f; }
  __syncthreads();
  for (int t = 0; t < CC; ++t) {
    // gates1 = G1[t] + h1 @ Whh1^T   (M=32,N=128 -> 16 tiles, waves 0..15)
    if (warp < 16) {
      int mt = warp & 1, nt = warp >> 1;
      int brow = mt * 16 + ln;
      int col = nt * 16 + ln;
      bf16x16 a, b;
#pragma unroll
      for (int p = 0; p < 16; ++p) {
        a[p] = h1s[brow * H1D + akidx(hi, p)];
        b[p] = (__bf16)Whh1[col * H1D + bkidx(hi, p)];
      }
      f32x8 c;
#pragma unroll
      for (int r = 0; r < 8; ++r) c[r] = G1[(t * NBATCH + mt * 16 + r + 8 * hi) * (4 * H1D) + col];
      c = __builtin_amdgcn_wmma_f32_16x16x32_bf16(false, a, false, b, (short)0, c, false, false);
#pragma unroll
      for (int r = 0; r < 8; ++r) g2s[(mt * 16 + r + 8 * hi) * (4 * H1D) + col] = (__bf16)c[r];
    }
    __syncthreads();
    // LSTM1 pointwise: one thread per (b, hd)
    {
      int bb = tid >> 5, hd = tid & 31;
      int gb = bb * (4 * H1D);
      float iv = (float)g2s[gb + hd];
      float fv = (float)g2s[gb + H1D + hd];
      float gv = (float)g2s[gb + 2 * H1D + hd];
      float ov = (float)g2s[gb + 3 * H1D + hd];
      float cnew = sigmoidf(fv) * c1s[tid] + sigmoidf(iv) * tanhf(gv);
      c1s[tid] = cnew;
      h1s[tid] = (__bf16)(sigmoidf(ov) * tanhf(cnew));
    }
    __syncthreads();
    // gates2 = h1 @ Wih2^T + h2 @ Whh2^T + b   (M=32,N=512 -> 64 tiles, 2 per wave)
#pragma unroll
    for (int q = 0; q < 2; ++q) {
      int tt = warp + q * 32;
      int mt = tt & 1, nt = tt >> 1;
      int brow = mt * 16 + ln;
      int col = nt * 16 + ln;                // 0..511
      float bias = bih2[col] + bhh2[col];
      f32x8 c;
#pragma unroll
      for (int r = 0; r < 8; ++r) c[r] = bias;
      {
        bf16x16 a, b;
#pragma unroll
        for (int p = 0; p < 16; ++p) {
          a[p] = h1s[brow * H1D + akidx(hi, p)];
          b[p] = (__bf16)Wih2[col * H1D + bkidx(hi, p)];
        }
        c = __builtin_amdgcn_wmma_f32_16x16x32_bf16(false, a, false, b, (short)0, c, false, false);
      }
#pragma unroll
      for (int kt = 0; kt < 4; ++kt) {
        bf16x16 a, b;
#pragma unroll
        for (int p = 0; p < 16; ++p) {
          a[p] = h2s[brow * H2D + kt * 32 + akidx(hi, p)];
          b[p] = (__bf16)Whh2[col * H2D + kt * 32 + bkidx(hi, p)];
        }
        c = __builtin_amdgcn_wmma_f32_16x16x32_bf16(false, a, false, b, (short)0, c, false, false);
      }
#pragma unroll
      for (int r = 0; r < 8; ++r) g2s[(mt * 16 + r + 8 * hi) * (4 * H2D) + col] = (__bf16)c[r];
    }
    __syncthreads();
    // LSTM2 pointwise: 4096 states, 4 per thread
#pragma unroll
    for (int q = 0; q < 4; ++q) {
      int s = tid + q * 1024;
      int bb = s >> 7, hd = s & 127;
      int gb = bb * (4 * H2D);
      float iv = (float)g2s[gb + hd];
      float fv = (float)g2s[gb + H2D + hd];
      float gv = (float)g2s[gb + 2 * H2D + hd];
      float ov = (float)g2s[gb + 3 * H2D + hd];
      float cnew = sigmoidf(fv) * c2s[s] + sigmoidf(iv) * tanhf(gv);
      c2s[s] = cnew;
      h2s[s] = (__bf16)(sigmoidf(ov) * tanhf(cnew));
    }
    __syncthreads();
  }
  for (int s = tid; s < NBATCH * H2D; s += 1024) h2out[s] = (float)h2s[s];
}

// ---------------- head: y = h2_last @ W_lin^T + b_lin  (M=32, N=2000, K=128)
// flat y[b*2000 + j*2 + p] == d_out[(b*1000+j)*2 + p]  -> direct store
__global__ void k_head(const float* __restrict__ h2out, const float* __restrict__ Wlin,
                       const float* __restrict__ blin, float* __restrict__ out) {
  int lane = threadIdx.x & 31, warp = threadIdx.x >> 5;
  int gw = blockIdx.x * 8 + warp;
  if (gw >= 250) return;                     // 2 x 125 tiles
  int mt = gw & 1, nt = gw >> 1;
  int hi = lane >> 4, ln = lane & 15;
  int brow = mt * 16 + ln;
  int col = nt * 16 + ln;                    // < 2000
  f32x8 c;
#pragma unroll
  for (int r = 0; r < 8; ++r) c[r] = blin[col];
#pragma unroll
  for (int kt = 0; kt < 4; ++kt) {
    bf16x16 a, b;
#pragma unroll
    for (int p = 0; p < 16; ++p) {
      a[p] = (__bf16)h2out[brow * H2D + kt * 32 + akidx(hi, p)];
      b[p] = (__bf16)Wlin[(long)col * H2D + kt * 32 + bkidx(hi, p)];
    }
    c = __builtin_amdgcn_wmma_f32_16x16x32_bf16(false, a, false, b, (short)0, c, false, false);
  }
#pragma unroll
  for (int r = 0; r < 8; ++r) {
    int m = mt * 16 + r + 8 * hi;
    out[(long)m * (NNODE * NPRED) + col] = c[r];
  }
}

extern "C" void kernel_launch(void* const* d_in, const int* in_sizes, int n_in,
                              void* d_out, int out_size, void* d_ws, size_t ws_size,
                              hipStream_t stream) {
  (void)in_sizes; (void)n_in; (void)out_size; (void)ws_size;
  const float* x     = (const float*)d_in[0];
  const int*   src   = (const int*)d_in[1];
  const int*   dst   = (const int*)d_in[2];
  const float* Wg    = (const float*)d_in[4];
  const float* a_src = (const float*)d_in[5];
  const float* a_dst = (const float*)d_in[6];
  const float* b_gat = (const float*)d_in[7];
  const float* Wih1  = (const float*)d_in[8];
  const float* Whh1  = (const float*)d_in[9];
  const float* bih1  = (const float*)d_in[10];
  const float* bhh1  = (const float*)d_in[11];
  const float* Wih2  = (const float*)d_in[12];
  const float* Whh2  = (const float*)d_in[13];
  const float* bih2  = (const float*)d_in[14];
  const float* bhh2  = (const float*)d_in[15];
  const float* Wlin  = (const float*)d_in[16];
  const float* blin  = (const float*)d_in[17];
  float* out = (float*)d_out;

  char* ws = (char*)d_ws;
  size_t off = 0;
  float*    xlin   = (float*)(ws + off);    off += (size_t)NTOT * NH * CC * 4;   // 32.77 MB
  float*    asrc   = (float*)(ws + off);    off += (size_t)NTOT * NH * 4;
  float*    adst   = (float*)(ws + off);    off += (size_t)NTOT * NH * 4;
  unsigned* menc   = (unsigned*)(ws + off); off += (size_t)NTOT * NH * 4;
  float*    denom  = (float*)(ws + off);    off += (size_t)NTOT * NH * 4;
  float*    esc    = (float*)(ws + off);    off += (size_t)NE * NH * 4;          // 12.29 MB
  float*    gatout = (float*)(ws + off);    off += (size_t)NTOT * CC * 4;        // 4.10 MB
  float*    G1     = (float*)(ws + off);    off += (size_t)NBATCH * CC * 4 * H1D * 4;
  float*    h2out  = (float*)(ws + off);    off += (size_t)NBATCH * H2D * 4;

  k_init<<<(NTOT * CC + 255) / 256, 256, 0, stream>>>(gatout, b_gat, menc, denom);
  k_gat_gemm<<<(2000 * 16) / 8, 256, 0, stream>>>(x, Wg, xlin);      // 32000 waves
  k_attn<<<(NTOT * NH + 255) / 256, 256, 0, stream>>>(xlin, a_src, a_dst, asrc, adst);
  k_edge_score<<<(NE + 255) / 256, 256, 0, stream>>>(src, dst, asrc, adst, esc, menc);
  k_edge_denom<<<(NE + 255) / 256, 256, 0, stream>>>(dst, menc, esc, denom);
  k_edge_agg<<<(NE + 255) / 256, 256, 0, stream>>>(src, dst, esc, denom, xlin, gatout);
  k_lstm_pre<<<64, 256, 0, stream>>>(gatout, Wih1, bih1, bhh1, G1);  // 512 waves, async-LDS staged
  k_lstm_rec<<<1, 1024, 0, stream>>>(G1, Whh1, Wih2, Whh2, bih2, bhh2, h2out);
  k_head<<<32, 256, 0, stream>>>(h2out, Wlin, blin, out);            // 250 waves used
}
